// NodeModel_68453188763945
// MI455X (gfx1250) — matrix operand order repaired and checked by reading, not compiled, not run on hardware
//
#include <hip/hip_runtime.h>

typedef __attribute__((ext_vector_type(16))) __bf16 v16bf;
typedef __attribute__((ext_vector_type(8)))  __bf16 v8bf;
typedef __attribute__((ext_vector_type(4)))  __bf16 v4bf;
typedef __attribute__((ext_vector_type(8)))  float  v8f;
typedef __attribute__((ext_vector_type(4)))  float  v4f;

constexpr int N_NODES = 100000;
constexpr int N_EDGES = 1600000;
constexpr int D_NODE  = 128;
constexpr int D_EDGE  = 128;
constexpr int D_GLOB  = 16;
constexpr int D_HID   = 256;
constexpr int D_IN    = D_NODE + D_EDGE + D_GLOB;  // 272
constexpr int D_INP   = 288;                       // padded to 9 K-tiles of 32

constexpr size_t AGG_BYTES = (size_t)N_NODES * D_NODE * 4;
constexpr int    W1_FRAGS  = 9 * 16;   // ktiles * ntiles
constexpr int    W2_FRAGS  = 8 * 8;
constexpr size_t W1P_BYTES = (size_t)W1_FRAGS * 32 * 16 * 2;  // frag*lane*16 bf16

__device__ __forceinline__ v16bf cat16(v8bf a, v8bf b) {
    return __builtin_shufflevector(a, b, 0,1,2,3,4,5,6,7,8,9,10,11,12,13,14,15);
}

// ---------------------------------------------------------------- zero agg
__global__ void zero_f4(v4f* __restrict__ p, int n4) {
    int i = blockIdx.x * blockDim.x + threadIdx.x;
    int stride = gridDim.x * blockDim.x;
    v4f z = {0.f, 0.f, 0.f, 0.f};
    for (; i < n4; i += stride) p[i] = z;
}

// ------------------------------------------------- pack W1 -> bf16 B-fragments
// B-fragment layout (16-bit, 32x16): lanes 0-15 hold K=kt*32+0..15, col n=lane&15;
// lanes 16-31 hold K=kt*32+16..31. 16 bf16 per lane, contiguous in K.
__global__ void pack_w1(const float* __restrict__ W1, __bf16* __restrict__ W1p) {
    int t = blockIdx.x * blockDim.x + threadIdx.x;
    if (t >= W1_FRAGS * 32) return;
    int lane = t & 31;
    int f    = t >> 5;
    int kt   = f >> 4;          // 0..8
    int nt   = f & 15;          // 0..15
    int n    = nt * 16 + (lane & 15);
    int kb   = kt * 32 + ((lane >> 4) & 1) * 16;
    __bf16 e[16];
#pragma unroll
    for (int j = 0; j < 16; ++j) {
        int k = kb + j;
        e[j] = (k < D_IN) ? (__bf16)W1[(size_t)k * D_HID + n] : (__bf16)0.0f;
    }
    v8bf lo, hi;
#pragma unroll
    for (int j = 0; j < 8; ++j) { lo[j] = e[j]; hi[j] = e[8 + j]; }
    v8bf* dst = (v8bf*)(W1p + (size_t)t * 16);
    dst[0] = lo;
    dst[1] = hi;
}

__global__ void pack_w2(const float* __restrict__ W2, __bf16* __restrict__ W2p) {
    int t = blockIdx.x * blockDim.x + threadIdx.x;
    if (t >= W2_FRAGS * 32) return;
    int lane = t & 31;
    int f    = t >> 5;
    int kt   = f >> 3;          // 0..7
    int nt   = f & 7;           // 0..7
    int n    = nt * 16 + (lane & 15);
    int kb   = kt * 32 + ((lane >> 4) & 1) * 16;
    __bf16 e[16];
#pragma unroll
    for (int j = 0; j < 16; ++j)
        e[j] = (__bf16)W2[(size_t)(kb + j) * D_NODE + n];
    v8bf lo, hi;
#pragma unroll
    for (int j = 0; j < 8; ++j) { lo[j] = e[j]; hi[j] = e[8 + j]; }
    v8bf* dst = (v8bf*)(W2p + (size_t)t * 16);
    dst[0] = lo;
    dst[1] = hi;
}

// ------------------------------------------------------- edge scatter-sum
// One wave per edge: lane NT-loads float4 of edge_attr (single-use 819MB
// stream, keep out of L2), 4 global fp32 atomics into L2-resident agg.
__global__ __launch_bounds__(256) void scatter_edges(
    const float* __restrict__ ea, const int* __restrict__ col,
    float* __restrict__ agg) {
    int lane = threadIdx.x & 31;
    int e    = blockIdx.x * 8 + (threadIdx.x >> 5);
    if (e >= N_EDGES) return;
    int dst  = col[e];
    v4f v = __builtin_nontemporal_load((const v4f*)(ea + (size_t)e * D_EDGE) + lane);
    float* p = agg + (size_t)dst * D_NODE + lane * 4;
    unsafeAtomicAdd(p + 0, v.x);
    unsafeAtomicAdd(p + 1, v.y);
    unsafeAtomicAdd(p + 2, v.z);
    unsafeAtomicAdd(p + 3, v.w);
}

// ------------------------------------------------------- fused MLP (WMMA)
__global__ __launch_bounds__(128) void fused_mlp(
    const float* __restrict__ x,   const float* __restrict__ agg,
    const float* __restrict__ u,   const int* __restrict__ batch,
    const __bf16* __restrict__ W1p, const float* __restrict__ b1,
    const __bf16* __restrict__ W2p, const float* __restrict__ b2,
    float* __restrict__ out) {
    __shared__ __attribute__((aligned(16))) __bf16 smem[4 * 16 * D_INP]; // 36,864 B
    const int lane = threadIdx.x & 31;
    const int wave = threadIdx.x >> 5;
    __bf16* buf   = smem + wave * 16 * D_INP;   // private 9,216 B per wave
    const int m0  = blockIdx.x * 64 + wave * 16;
    const int hi  = lane >> 4;      // 0 or 1
    const int r0  = lane & 15;

    // ---- stage [x | agg | u[batch] | 0-pad] rows into LDS as bf16 ----
#pragma unroll 4
    for (int i = 0; i < 16; ++i) {
        int node = m0 + i; if (node >= N_NODES) node = N_NODES - 1;
        v4f vx = ((const v4f*)(x   + (size_t)node * D_NODE))[lane];
        v4f va = ((const v4f*)(agg + (size_t)node * D_NODE))[lane];
        v4bf tx, ta;
        tx[0] = (__bf16)vx.x; tx[1] = (__bf16)vx.y; tx[2] = (__bf16)vx.z; tx[3] = (__bf16)vx.w;
        ta[0] = (__bf16)va.x; ta[1] = (__bf16)va.y; ta[2] = (__bf16)va.z; ta[3] = (__bf16)va.w;
        *(v4bf*)(buf + i * D_INP + lane * 4)          = tx;
        *(v4bf*)(buf + i * D_INP + D_NODE + lane * 4) = ta;
    }
#pragma unroll
    for (int i0 = 0; i0 < 16; i0 += 8) {
        int i = i0 + (lane >> 2);
        int q = lane & 3;
        int node = m0 + i; if (node >= N_NODES) node = N_NODES - 1;
        int g = batch[node];
        v4f vu = ((const v4f*)u)[g * 4 + q];
        v4bf tu;
        tu[0] = (__bf16)vu.x; tu[1] = (__bf16)vu.y; tu[2] = (__bf16)vu.z; tu[3] = (__bf16)vu.w;
        *(v4bf*)(buf + i * D_INP + (D_NODE + D_EDGE) + q * 4) = tu;
        v4bf z = {};
        *(v4bf*)(buf + i * D_INP + D_IN + q * 4) = z;   // pad K 272..287
    }
    __builtin_amdgcn_wave_barrier();
    asm volatile("s_wait_dscnt 0x0" ::: "memory");

    // ---- GEMM1: h[16x256] = relu(in[16x288] @ W1 + b1), bf16 WMMA ----
    const v8f zv = {0.f,0.f,0.f,0.f,0.f,0.f,0.f,0.f};
    v8f acc[16];
#pragma unroll
    for (int nt = 0; nt < 16; ++nt) acc[nt] = zv;

    const v8bf* w1v = (const v8bf*)W1p;
    for (int kt = 0; kt < 9; ++kt) {
        // A layout (16-bit 16x32): e0..15 = K kb..kb+15, kb=hi*16
        const __bf16* ap = buf + r0 * D_INP + kt * 32 + hi * 16;
        v16bf afrag = cat16(*(const v8bf*)ap, *(const v8bf*)(ap + 8));
        // software-pipelined B fragments: prefetch nt+1 before wmma(nt)
        const v8bf* bp0 = w1v + (size_t)(kt * 16) * 64 + lane * 2;
        v16bf bnext = cat16(bp0[0], bp0[1]);
#pragma unroll
        for (int nt = 0; nt < 16; ++nt) {
            v16bf bcur = bnext;
            if (nt < 15) {
                const v8bf* bq = w1v + (size_t)(kt * 16 + nt + 1) * 64 + lane * 2;
                bnext = cat16(bq[0], bq[1]);
            }
            acc[nt] = __builtin_amdgcn_wmma_f32_16x16x32_bf16(
                false, afrag, false, bcur, (short)0, acc[nt], false, false);
        }
    }

    // ---- bias + relu, h -> LDS (reuse buf; all A fragments consumed) ----
    __builtin_amdgcn_wave_barrier();
#pragma unroll
    for (int nt = 0; nt < 16; ++nt) {
        int   c    = nt * 16 + r0;
        float bias = b1[c];
#pragma unroll
        for (int r = 0; r < 8; ++r) {           // C layout: VGPR r -> row r + hi*8
            float v = acc[nt][r] + bias;
            v = v > 0.f ? v : 0.f;
            buf[(r + hi * 8) * D_INP + c] = (__bf16)v;
        }
    }
    __builtin_amdgcn_wave_barrier();
    asm volatile("s_wait_dscnt 0x0" ::: "memory");

    // ---- GEMM2: out[16x128] = h[16x256] @ W2 ----
    v8f acc2[8];
#pragma unroll
    for (int nt = 0; nt < 8; ++nt) acc2[nt] = zv;

    const v8bf* w2v = (const v8bf*)W2p;
    for (int kt = 0; kt < 8; ++kt) {
        const __bf16* ap = buf + r0 * D_INP + kt * 32 + hi * 16;
        v16bf afrag = cat16(*(const v8bf*)ap, *(const v8bf*)(ap + 8));
        const v8bf* bp0 = w2v + (size_t)(kt * 8) * 64 + lane * 2;
        v16bf bnext = cat16(bp0[0], bp0[1]);
#pragma unroll
        for (int nt = 0; nt < 8; ++nt) {
            v16bf bcur = bnext;
            if (nt < 7) {
                const v8bf* bq = w2v + (size_t)(kt * 8 + nt + 1) * 64 + lane * 2;
                bnext = cat16(bq[0], bq[1]);
            }
            acc2[nt] = __builtin_amdgcn_wmma_f32_16x16x32_bf16(
                false, afrag, false, bcur, (short)0, acc2[nt], false, false);
        }
    }

    // ---- epilogue: stage (acc2 + b2) through LDS as f32, then coalesced
    //      b128 stores with residual x added; NT store (write-once stream) ----
    float* fb = (float*)buf;                    // reuse region: 16 x 128 f32 = 8 KB
    __builtin_amdgcn_wave_barrier();
#pragma unroll
    for (int nt = 0; nt < 8; ++nt) {
        int   c    = nt * 16 + r0;
        float bias = b2[c];
#pragma unroll
        for (int r = 0; r < 8; ++r)
            fb[(r + hi * 8) * D_NODE + c] = acc2[nt][r] + bias;
    }
    __builtin_amdgcn_wave_barrier();
    asm volatile("s_wait_dscnt 0x0" ::: "memory");

#pragma unroll 4
    for (int i = 0; i < 16; ++i) {
        int node = m0 + i;
        if (node < N_NODES) {
            v4f vr = ((const v4f*)fb)[i * 32 + lane];
            v4f vx = ((const v4f*)(x + (size_t)node * D_NODE))[lane];
            v4f o  = {vr.x + vx.x, vr.y + vx.y, vr.z + vx.z, vr.w + vx.w};
            __builtin_nontemporal_store(o, (v4f*)(out + (size_t)node * D_NODE) + lane);
        }
    }
}

// ---------------------------------------------------------------- launcher
extern "C" void kernel_launch(void* const* d_in, const int* in_sizes, int n_in,
                              void* d_out, int out_size, void* d_ws, size_t ws_size,
                              hipStream_t stream) {
    const float* x  = (const float*)d_in[0];
    const int*   ei = (const int*)  d_in[1];   // [2, N_EDGES]; row 1 = dst
    const float* ea = (const float*)d_in[2];
    const float* u  = (const float*)d_in[3];
    const int*   bt = (const int*)  d_in[4];
    const float* W1 = (const float*)d_in[5];
    const float* b1 = (const float*)d_in[6];
    const float* W2 = (const float*)d_in[7];
    const float* b2 = (const float*)d_in[8];
    float* out = (float*)d_out;

    char*   ws  = (char*)d_ws;
    float*  agg = (float*)ws;
    __bf16* W1p = (__bf16*)(ws + AGG_BYTES);
    __bf16* W2p = (__bf16*)(ws + AGG_BYTES + W1P_BYTES);

    const int n4 = N_NODES * D_NODE / 4;
    zero_f4<<<4096, 256, 0, stream>>>((v4f*)agg, n4);
    pack_w1<<<(W1_FRAGS * 32 + 255) / 256, 256, 0, stream>>>(W1, W1p);
    pack_w2<<<(W2_FRAGS * 32 + 255) / 256, 256, 0, stream>>>(W2, W2p);
    scatter_edges<<<N_EDGES / 8, 256, 0, stream>>>(ea, ei + N_EDGES, agg);
    fused_mlp<<<(N_NODES + 63) / 64, 128, 0, stream>>>(
        x, agg, u, bt, W1p, b1, W2p, b2, out);
}
